// Decoder_7928509628599
// MI455X (gfx1250) — matrix operand order repaired and checked by reading, not compiled, not run on hardware
//
#include <hip/hip_runtime.h>
#include <hip/hip_bf16.h>
#include <cstdint>
#include <cmath>

typedef __attribute__((ext_vector_type(16))) __bf16 v16bf;
typedef __attribute__((ext_vector_type(8)))  __bf16 v8bf;
typedef __attribute__((ext_vector_type(8)))  float  v8f;

#define S_STEPS 400
#define BATCH   64
#define T_FR    1200
#define KG      1792      // 256 (prenet) + 512 (memory) + 1024 (recurrent)
#define NG      4096
#define HID     1024
#define PREW    256
#define FRAME   240

// ---------------- deterministic dropout hash ----------------
__device__ __forceinline__ float hash01(uint32_t x) {
    x ^= x >> 17; x *= 0xed5ad4bbu;
    x ^= x >> 11; x *= 0xac4c1b51u;
    x ^= x >> 15; x *= 0x31848babu;
    x ^= x >> 14;
    return (float)(x >> 8) * (1.0f / 16777216.0f);
}
__device__ __forceinline__ float sigf(float x) { return 1.0f / (1.0f + __expf(-x)); }

// ---------------- WMMA fragment helpers (bf16, 16x16x32) ----------------
// A-matrix 16x32 (MxK), row-major source with leading dim lda.
// lane<16: M=lane,    K = k0 + {0..7, 16..23}
// lane>=16: M=lane-16, K = k0 + {8..15, 24..31}
__device__ __forceinline__ v16bf load_a(const __bf16* __restrict__ A, int lda, int m0, int k0) {
    int lane = threadIdx.x & 31;
    const __bf16* p = A + (size_t)(m0 + (lane & 15)) * lda + k0 + ((lane & 16) ? 8 : 0);
    v8bf lo = *(const v8bf*)p;          // 16B aligned: lda,k0 multiples of 32, offset 0/8 elems
    v8bf hi = *(const v8bf*)(p + 16);
    v16bf a;
#pragma unroll
    for (int e = 0; e < 8; ++e) { a[e] = lo[e]; a[e + 8] = hi[e]; }
    return a;
}
// B-matrix 32x16 (KxN) sourced from row-major W[N][K] (i.e. B = W^T), leading dim ldb.
// lane<16: N=lane,    K = k0 + {0..7, 16..23} ; lane>=16: N=lane-16, K = k0 + {8..15, 24..31}
__device__ __forceinline__ v16bf load_b(const __bf16* __restrict__ W, int ldb, int n0, int k0) {
    int lane = threadIdx.x & 31;
    const __bf16* p = W + (size_t)(n0 + (lane & 15)) * ldb + k0 + ((lane & 16) ? 8 : 0);
    v8bf lo = *(const v8bf*)p;
    v8bf hi = *(const v8bf*)(p + 16);
    v16bf b;
#pragma unroll
    for (int e = 0; e < 8; ++e) { b[e] = lo[e]; b[e + 8] = hi[e]; }
    return b;
}
__device__ __forceinline__ v8f wmma_bf16(v16bf a, v16bf b, v8f c) {
    return __builtin_amdgcn_wmma_f32_16x16x32_bf16(false, a, false, b, (short)0, c, false, false);
}

// ---------------- one-time prep kernels ----------------
// Concatenate w_ih[4096x768] | w_hh[4096x1024] -> bf16 Wg[4096x1792]
__global__ void prep_wg(const float* __restrict__ wih, const float* __restrict__ whh,
                        __bf16* __restrict__ Wg) {
    int idx = blockIdx.x * blockDim.x + threadIdx.x;          // 4096*1792
    int k = idx % KG, n = idx / KG;
    float v = (k < 768) ? wih[(size_t)n * 768 + k] : whh[(size_t)n * 1024 + (k - 768)];
    Wg[idx] = (__bf16)v;
}
__global__ void prep_small(const float* __restrict__ w1, const float* __restrict__ w2,
                           const float* __restrict__ pw, const float* __restrict__ mem,
                           const float* __restrict__ bih, const float* __restrict__ bhh,
                           __bf16* __restrict__ w1p, __bf16* __restrict__ w2p,
                           __bf16* __restrict__ pwb, __bf16* __restrict__ memb,
                           float* __restrict__ bsum) {
    int idx = blockIdx.x * blockDim.x + threadIdx.x;          // covers 240*1024
    if (idx < 256 * 256) {
        int k = idx & 255, h = idx >> 8;
        w1p[idx] = (__bf16)(k < FRAME ? w1[h * FRAME + k] : 0.0f);   // K padded 240->256
        w2p[idx] = (__bf16)w2[idx];
    }
    if (idx < FRAME * HID) pwb[idx] = (__bf16)pw[idx];
    if (idx < BATCH * 512) memb[idx] = (__bf16)mem[idx];
    if (idx < NG)          bsum[idx] = bih[idx] + bhh[idx];
}
// decoder_inputs [B,80,1200] -> dib [S][B][256] bf16 (zero-padded K, zero go-frame)
__global__ void build_dib(const float* __restrict__ din, __bf16* __restrict__ dib) {
    int idx = blockIdx.x * blockDim.x + threadIdx.x;          // 400*64*256
    int j = idx & 255, b = (idx >> 8) & 63, s = idx >> 14;
    float v = 0.0f;
    if (j < FRAME && s > 0) {
        int m = j % 80, t = 3 * (s - 1) + j / 80;
        v = din[((size_t)b * 80 + m) * T_FR + t];
    }
    dib[idx] = (__bf16)v;
}
__global__ void init_state(__bf16* __restrict__ hb, __bf16* __restrict__ hnb,
                           float* __restrict__ dcf) {
    int idx = blockIdx.x * blockDim.x + threadIdx.x;          // 64*1024
    hb[idx] = (__bf16)0.0f; hnb[idx] = (__bf16)0.0f; dcf[idx] = 0.0f;
}

// ---------------- prenet: [S*64 x 256] x [256 -> 256], ReLU + dropout(0.5) ----------------
// Each wave: one 16(M) x 64(N) strip -> 4 accumulators; A fragment reused 4x;
// all B fragments fetched before the WMMA burst so waits are partial, not full.
__global__ void prenet_gemm(const __bf16* __restrict__ A, const __bf16* __restrict__ W,
                            __bf16* __restrict__ O, uint32_t seed) {
    int wid = blockIdx.x * (blockDim.x >> 5) + (threadIdx.x >> 5);
    if (wid >= S_STEPS * 16) return;                          // uniform per wave
    int s = wid >> 4, rem = wid & 15;
    int m0 = (rem >> 2) << 4;          // 4 M-tiles
    int ng = (rem & 3) << 6;           // 4 N-groups of 64
    const __bf16* As = A + (size_t)s * BATCH * PREW;
    v8f acc[4] = {};
#pragma unroll
    for (int kc = 0; kc < PREW; kc += 32) {
        v16bf a = load_a(As, PREW, m0, kc);
        v16bf bfr[4];
#pragma unroll
        for (int t = 0; t < 4; ++t) bfr[t] = load_b(W, PREW, ng + t * 16, kc);
#pragma unroll
        for (int t = 0; t < 4; ++t) acc[t] = wmma_bf16(a, bfr[t], acc[t]);
    }
    int lane = threadIdx.x & 31;
    int mb = m0 + ((lane & 16) ? 8 : 0);
#pragma unroll
    for (int t = 0; t < 4; ++t) {
        int n = ng + t * 16 + (lane & 15);
#pragma unroll
        for (int r = 0; r < 8; ++r) {
            int m = mb + r;
            float v = acc[t][r]; v = v > 0.0f ? v : 0.0f;
            uint32_t li = ((uint32_t)s * 64u + (uint32_t)m) * 256u + (uint32_t)n;
            v = (hash01(li * 2654435761u + seed) < 0.5f) ? v * 2.0f : 0.0f;  // p_keep = 0.5
            O[(size_t)s * BATCH * PREW + (size_t)m * PREW + n] = (__bf16)v;
        }
    }
}

// ---------------- per-step gates GEMM: [64 x 1792] x Wg^T -> G[64 x 4096] + bias ----------------
// Each wave: 16(M) x 64(N) strip, 4 independent accumulators.
// A fragment (prenet|memory|state concat) fetched once per K-step, reused 4x;
// B fragments batched ahead of the WMMA burst for load/compute overlap.
__global__ void gates_kernel(const __bf16* __restrict__ pre_s, const __bf16* __restrict__ memb,
                             const __bf16* __restrict__ hb, const __bf16* __restrict__ Wg,
                             const float* __restrict__ bsum, float* __restrict__ G) {
    int wid = blockIdx.x * (blockDim.x >> 5) + (threadIdx.x >> 5);   // 256 waves
    int m0 = (wid >> 6) << 4;          // 4 M-tiles
    int ng = (wid & 63) << 6;          // 64 N-groups of 64
    v8f acc[4] = {};
#pragma unroll 2
    for (int kc = 0; kc < KG; kc += 32) {
        v16bf a;
        if (kc < 256)      a = load_a(pre_s, 256, m0, kc);            // prenet output
        else if (kc < 768) a = load_a(memb, 512, m0, kc - 256);       // memory
        else               a = load_a(hb, HID, m0, kc - 768);         // recurrent (swapped: c_new)
        v16bf bfr[4];
#pragma unroll
        for (int t = 0; t < 4; ++t) bfr[t] = load_b(Wg, KG, ng + t * 16, kc);
#pragma unroll
        for (int t = 0; t < 4; ++t) acc[t] = wmma_bf16(a, bfr[t], acc[t]);
    }
    int lane = threadIdx.x & 31;
    int mb = m0 + ((lane & 16) ? 8 : 0);
#pragma unroll
    for (int t = 0; t < 4; ++t) {
        int n = ng + t * 16 + (lane & 15);
        float bia = bsum[n];
#pragma unroll
        for (int r = 0; r < 8; ++r) G[(size_t)(mb + r) * NG + n] = acc[t][r] + bia;
    }
}

// ---------------- elementwise LSTM cell (with the reference's swapped carry) ----------------
__global__ void cell_kernel(const float* __restrict__ G, float* __restrict__ dcf,
                            __bf16* __restrict__ hb, __bf16* __restrict__ hnb) {
    int idx = blockIdx.x * blockDim.x + threadIdx.x;          // 64*1024
    int b = idx >> 10, u = idx & 1023;
    const float* g = G + (size_t)b * NG;
    float i_ = g[u], f_ = g[1024 + u], gg = g[2048 + u], o_ = g[3072 + u];
    float c_new = sigf(f_) * dcf[idx] + sigf(i_) * tanhf(gg);
    float h_new = sigf(o_) * tanhf(c_new);
    hb[idx]  = (__bf16)c_new;   // next step's recurrent matmul input (carry swap)
    hnb[idx] = (__bf16)h_new;   // projection input
    dcf[idx] = h_new;           // next step's leak term (carry swap)
}

// ---------------- projection: [64 x 1024] x proj_w^T -> mel, scattered to [B,80,T] ----------------
__global__ void proj_kernel(const __bf16* __restrict__ hnb, const __bf16* __restrict__ pwb,
                            const float* __restrict__ pb, float* __restrict__ out, int s) {
    int wid = blockIdx.x * (blockDim.x >> 5) + (threadIdx.x >> 5);
    if (wid >= 4 * 15) return;                                // uniform per wave
    int m0 = (wid / 15) * 16, n0 = (wid % 15) * 16;
    v8f acc = {};
#pragma unroll 8
    for (int kc = 0; kc < HID; kc += 32) {
        acc = wmma_bf16(load_a(hnb, HID, m0, kc), load_b(pwb, HID, n0, kc), acc);
    }
    int lane = threadIdx.x & 31;
    int j = n0 + (lane & 15), mb = m0 + ((lane & 16) ? 8 : 0);
    float bias = pb[j];
    int m80 = j % 80, tq = j / 80;
#pragma unroll
    for (int r = 0; r < 8; ++r) {
        int b = mb + r;
        out[((size_t)b * 80 + m80) * T_FR + 3 * s + tq] = acc[r] + bias;
    }
}

// ---------------- host orchestration ----------------
extern "C" void kernel_launch(void* const* d_in, const int* in_sizes, int n_in,
                              void* d_out, int out_size, void* d_ws, size_t ws_size,
                              hipStream_t stream) {
    const float* memory = (const float*)d_in[0];
    const float* dec_in = (const float*)d_in[1];
    // d_in[2] = memory_lengths (unused by the reference)
    const float* w1  = (const float*)d_in[3];
    const float* w2  = (const float*)d_in[4];
    const float* wih = (const float*)d_in[5];
    const float* whh = (const float*)d_in[6];
    const float* bih = (const float*)d_in[7];
    const float* bhh = (const float*)d_in[8];
    const float* pw  = (const float*)d_in[9];
    const float* pb  = (const float*)d_in[10];
    float* out = (float*)d_out;

    char* base = (char*)d_ws;
    size_t off = 0;
    auto carve = [&](size_t bytes) { size_t o = off; off = (off + bytes + 255) & ~(size_t)255; return o; };
    __bf16* Wg   = (__bf16*)(base + carve((size_t)NG * KG * 2));          // 14.7 MB
    __bf16* w1p  = (__bf16*)(base + carve(256 * 256 * 2));
    __bf16* w2p  = (__bf16*)(base + carve(256 * 256 * 2));
    __bf16* pwb  = (__bf16*)(base + carve((size_t)FRAME * HID * 2));
    __bf16* memb = (__bf16*)(base + carve(BATCH * 512 * 2));
    float*  bsum = (float*) (base + carve(NG * 4));
    __bf16* dib  = (__bf16*)(base + carve((size_t)S_STEPS * BATCH * PREW * 2)); // 13.1 MB
    __bf16* h1   = (__bf16*)(base + carve((size_t)S_STEPS * BATCH * PREW * 2)); // 13.1 MB
    __bf16* pre  = (__bf16*)(base + carve((size_t)S_STEPS * BATCH * PREW * 2)); // 13.1 MB
    __bf16* hb   = (__bf16*)(base + carve(BATCH * HID * 2));
    __bf16* hnb  = (__bf16*)(base + carve(BATCH * HID * 2));
    float*  dcf  = (float*) (base + carve(BATCH * HID * 4));
    float*  G    = (float*) (base + carve((size_t)BATCH * NG * 4));       // 1 MB
    (void)ws_size; (void)in_sizes; (void)n_in; (void)out_size;

    // one-time prep
    prep_wg   <<<(NG * KG) / 256, 256, 0, stream>>>(wih, whh, Wg);
    prep_small<<<(FRAME * HID + 255) / 256, 256, 0, stream>>>(w1, w2, pw, memory, bih, bhh,
                                                              w1p, w2p, pwb, memb, bsum);
    build_dib <<<(S_STEPS * BATCH * PREW) / 256, 256, 0, stream>>>(dec_in, dib);
    init_state<<<(BATCH * HID) / 256, 256, 0, stream>>>(hb, hnb, dcf);

    // prenet: two dense WMMA GEMMs over all steps, ReLU + dropout (16 waves per step slab)
    prenet_gemm<<<(S_STEPS * 16) / 8, 256, 0, stream>>>(dib, w1p, h1, 0x9E3779B9u);
    prenet_gemm<<<(S_STEPS * 16) / 8, 256, 0, stream>>>(h1, w2p, pre, 0x85EBCA6Bu);

    // sequential decoder scan
    for (int s = 0; s < S_STEPS; ++s) {
        gates_kernel<<<32, 256, 0, stream>>>(pre + (size_t)s * BATCH * PREW,
                                             memb, hb, Wg, bsum, G);
        cell_kernel <<<(BATCH * HID) / 256, 256, 0, stream>>>(G, dcf, hb, hnb);
        proj_kernel <<<8, 256, 0, stream>>>(hnb, pwb, pb, out, s);
    }
}
